// Attention_8272107012450
// MI455X (gfx1250) — compile-verified, matrix operand order checked by reading, not run on hardware
//
#include <hip/hip_runtime.h>
#include <stdint.h>

// ---------------------------------------------------------------------------
// Mixed global/local attention for MI455X (gfx1250, wave32, WMMA 16x16x32 bf16)
//   b=8, h=16, n=1024, dk=64, GLOBAL_NUM=12, LOCAL_NUM=4
// One workgroup (4 waves, 128 threads) handles one (batch, head, 16-row qblock):
//   phase 1: S = scale * gate * (Q' K'^T) via v_wmma_f32_16x16x32_bf16 -> LDS
//   phase 2: row max / exp / sum over the 16x1024 LDS stripe (f32)
//   phase 3: write normalized P (the 570MB output, coalesced)
//   phase 4: O = P V via v_wmma_f32_16x16x32_bf16
// Kernel is templated on IS_LOCAL so the 12 global heads carry zero gate code
// and no runtime branches in the tile loop.
// ---------------------------------------------------------------------------

typedef __attribute__((ext_vector_type(16))) __bf16 v16bf;
typedef __attribute__((ext_vector_type(8)))  float  v8f;

#define N_CTX   1024
#define DK      64
#define NHEAD   16
#define NGLOB   12
#define NLOC    4
#define NBATCH  8
#define S_LD    1028          // padded LDS row stride (floats): 8*1028 % 64 == 32 -> conflict-free
#define SCALE   0.125f        // 1/sqrt(64)

#define OUT_SZ  ((size_t)NBATCH * NHEAD * N_CTX * DK)        // 8,388,608
#define PG_SZ   ((size_t)NBATCH * NGLOB * N_CTX * N_CTX)     // 100,663,296
#define LDS_FLOATS (16 * S_LD + 128 + 48)                    // S stripe + reduce + stats

static __device__ __forceinline__ v8f wmma_bf16(const v16bf& a, const v16bf& b, const v8f& c) {
    return __builtin_amdgcn_wmma_f32_16x16x32_bf16(
        /*neg_a=*/false, a, /*neg_b=*/false, b,
        /*c_mod=*/(short)0, c, /*reuse_a=*/false, /*reuse_b=*/false);
}

// Pack p[0..7] (+ optional add w[0..7], compile-time selected) as bf16 into
// vector elements LO..LO+7.
template <int LO, bool HASW>
static __device__ __forceinline__ void load8_add(v16bf& a, const float* __restrict__ p,
                                                 const float* __restrict__ w) {
    float4 x0 = *(const float4*)(p);
    float4 x1 = *(const float4*)(p + 4);
    if constexpr (HASW) {
        float4 w0 = *(const float4*)(w);
        float4 w1 = *(const float4*)(w + 4);
        x0.x += w0.x; x0.y += w0.y; x0.z += w0.z; x0.w += w0.w;
        x1.x += w1.x; x1.y += w1.y; x1.z += w1.z; x1.w += w1.w;
    }
    a[LO + 0] = (__bf16)x0.x; a[LO + 1] = (__bf16)x0.y;
    a[LO + 2] = (__bf16)x0.z; a[LO + 3] = (__bf16)x0.w;
    a[LO + 4] = (__bf16)x1.x; a[LO + 5] = (__bf16)x1.y;
    a[LO + 6] = (__bf16)x1.z; a[LO + 7] = (__bf16)x1.w;
}

// Same, from LDS, with a scalar rescale (1/rowsum folded into the bf16 pack).
template <int LO>
static __device__ __forceinline__ void load8_scale(v16bf& a, const float* p, float sc) {
    float4 x0 = *(const float4*)(p);
    float4 x1 = *(const float4*)(p + 4);
    a[LO + 0] = (__bf16)(x0.x * sc); a[LO + 1] = (__bf16)(x0.y * sc);
    a[LO + 2] = (__bf16)(x0.z * sc); a[LO + 3] = (__bf16)(x0.w * sc);
    a[LO + 4] = (__bf16)(x1.x * sc); a[LO + 5] = (__bf16)(x1.y * sc);
    a[LO + 6] = (__bf16)(x1.z * sc); a[LO + 7] = (__bf16)(x1.w * sc);
}

template <bool IS_LOCAL>
__global__ __launch_bounds__(128)
void attn_mixed_kernel(const float* __restrict__ query,
                       const float* __restrict__ key,
                       const float* __restrict__ value,
                       const float* __restrict__ abs_q_w,
                       const float* __restrict__ abs_k_w,
                       const float* __restrict__ rel_w,
                       const uint8_t* __restrict__ mask,
                       float* __restrict__ out,
                       float* __restrict__ p_out) {   // p_g for global, p_l for local
    extern __shared__ float lds[];
    float* S      = lds;                  // 16 x S_LD score stripe
    float* red    = lds + 16 * S_LD;      // 16 x 8 partial reductions
    float* rowmax = red + 128;            // 16
    float* rowsum = rowmax + 16;          // 16
    float* invs   = rowsum + 16;          // 16

    const int tid   = threadIdx.x;
    const int wave  = tid >> 5;
    const int lane  = tid & 31;
    const int m     = lane & 15;          // row (A "M") or column ("N") index
    const int koff  = (lane < 16) ? 0 : 8;// K-chunk split per the 16-bit WMMA layout
    const int hi8   = (lane < 16) ? 0 : 8;// C/D: lanes 16-31 hold M = r+8

    const int q0    = blockIdx.x * 16;
    const int hIdx  = blockIdx.y;                       // index within head group
    const int head  = IS_LOCAL ? (NGLOB + hIdx) : hIdx; // absolute head
    const int batch = blockIdx.z;

    const size_t qkvBase = ((size_t)(batch * NHEAD + head)) << 16;   // *N_CTX*DK

    // ---------------- Phase 1: S = scale * gate * (Q' K'^T), masked ----------
    v16bf aq0, aq1;
    {
        const float* Qrow = query + qkvBase + (size_t)(q0 + m) * DK;
        const float* Wq   = abs_q_w + (size_t)hIdx * (N_CTX * DK) + (size_t)(q0 + m) * DK;
        load8_add<0, IS_LOCAL>(aq0, Qrow + koff,      Wq + koff);
        load8_add<8, IS_LOCAL>(aq0, Qrow + 16 + koff, Wq + 16 + koff);
        load8_add<0, IS_LOCAL>(aq1, Qrow + 32 + koff, Wq + 32 + koff);
        load8_add<8, IS_LOCAL>(aq1, Qrow + 48 + koff, Wq + 48 + koff);
    }

    const uint8_t* maskBH = mask + (size_t)batch * N_CTX * N_CTX;

    for (int i = 0; i < 16; ++i) {
        const int kbase = (wave + 4 * i) * 16;

        const float* Krow = key + qkvBase + (size_t)(kbase + m) * DK;
        const float* Wk   = abs_k_w + (size_t)hIdx * (N_CTX * DK) + (size_t)(kbase + m) * DK;

        // Prefetch the K row 256 rows (4 of this wave's tiles) ahead (gfx1250 global_prefetch_b8 path).
        if (i < 15) {
            __builtin_prefetch(Krow + 64 * 4 * DK, 0, 1);
            if constexpr (IS_LOCAL) __builtin_prefetch(Wk + 64 * 4 * DK, 0, 1);
        }

        v16bf bk0, bk1;
        load8_add<0, IS_LOCAL>(bk0, Krow + koff,      Wk + koff);
        load8_add<8, IS_LOCAL>(bk0, Krow + 16 + koff, Wk + 16 + koff);
        load8_add<0, IS_LOCAL>(bk1, Krow + 32 + koff, Wk + 32 + koff);
        load8_add<8, IS_LOCAL>(bk1, Krow + 48 + koff, Wk + 48 + koff);

        v8f c = {};
        c = wmma_bf16(aq0, bk0, c);
        c = wmma_bf16(aq1, bk1, c);

        // C layout: element r -> (M = r + hi8, N = lane&15)
        const int kk = kbase + m;
        #pragma unroll
        for (int r = 0; r < 8; ++r) {
            const int mm = r + hi8;
            const int qq = q0 + mm;
            float s = c[r] * SCALE;
            if constexpr (IS_LOCAL) {
                const float rw = rel_w[(size_t)(kk - qq + (N_CTX - 1)) * NLOC + hIdx];
                // sigmoid(rw/0.1) with fast reciprocal (v_rcp_f32, not precise div)
                s *= __builtin_amdgcn_rcpf(1.0f + __expf(-10.0f * rw));
            }
            if (!maskBH[(size_t)qq * N_CTX + kk]) s = -1.0e9f;
            S[mm * S_LD + kk] = s;
        }
    }
    __syncthreads();

    // ---------------- Phase 2: softmax stats over the 16x1024 stripe ---------
    const int row = tid >> 3;   // 16 rows, 8 threads each
    const int sub = tid & 7;
    {
        float mx = -3.0e38f;
        #pragma unroll 8
        for (int j = 0; j < 128; ++j)
            mx = fmaxf(mx, S[row * S_LD + sub + 8 * j]);
        red[row * 8 + sub] = mx;
    }
    __syncthreads();
    if (sub == 0) {
        float mx = red[row * 8];
        #pragma unroll
        for (int t = 1; t < 8; ++t) mx = fmaxf(mx, red[row * 8 + t]);
        rowmax[row] = mx;
    }
    __syncthreads();
    {
        const float rm = rowmax[row];
        float sm = 0.0f;
        #pragma unroll 8
        for (int j = 0; j < 128; ++j) {
            const int cidx = row * S_LD + sub + 8 * j;
            const float p = __expf(S[cidx] - rm);
            S[cidx] = p;                         // keep unnormalized exp in LDS
            sm += p;
        }
        red[row * 8 + sub] = sm;
    }
    __syncthreads();
    if (sub == 0) {
        float sm = 0.0f;
        #pragma unroll
        for (int t = 0; t < 8; ++t) sm += red[row * 8 + t];
        rowsum[row] = sm;
        invs[row]   = 1.0f / sm;                 // once per row: keep precise div
    }
    __syncthreads();

    // ---------------- Phase 3: write normalized P (bandwidth-critical) -------
    {
        const int NH = IS_LOCAL ? NLOC : NGLOB;
        float* pbase = p_out + ((size_t)(batch * NH + hIdx) * N_CTX + q0) * N_CTX;
        #pragma unroll 4
        for (int j = 0; j < 128; ++j) {
            const int flat = j * 128 + tid;          // consecutive lanes -> consecutive cols
            const int rr   = flat >> 10;
            const int cc   = flat & (N_CTX - 1);
            pbase[(size_t)rr * N_CTX + cc] = S[rr * S_LD + cc] * invs[rr];
        }
    }

    // ---------------- Phase 4: O = P V via WMMA ------------------------------
    {
        const int   ncb   = wave * 16;               // this wave's dk tile
        const float invm  = invs[m];                 // row scale folded into bf16 pack
        const float* Vb   = value + qkvBase;
        const int    col  = ncb + m;                 // B lane column (dk index)
        v8f acc = {};
        for (int kb = 0; kb < N_CTX; kb += 32) {
            v16bf pa, vb;
            const float* Srow = S + m * S_LD + kb + koff;
            load8_scale<0>(pa, Srow,      invm);
            load8_scale<8>(pa, Srow + 16, invm);
            #pragma unroll
            for (int e = 0; e < 8; ++e) {
                vb[e]     = (__bf16)Vb[(size_t)(kb + koff + e)      * DK + col];
                vb[8 + e] = (__bf16)Vb[(size_t)(kb + 16 + koff + e) * DK + col];
            }
            acc = wmma_bf16(pa, vb, acc);
        }
        float* obase = out + ((size_t)(batch * NHEAD + head) * N_CTX + q0) * DK;
        #pragma unroll
        for (int r = 0; r < 8; ++r)
            obase[(size_t)(r + hi8) * DK + ncb + m] = acc[r];
    }
}

extern "C" void kernel_launch(void* const* d_in, const int* in_sizes, int n_in,
                              void* d_out, int out_size, void* d_ws, size_t ws_size,
                              hipStream_t stream) {
    const float*   query   = (const float*)d_in[0];
    const float*   key     = (const float*)d_in[1];
    const float*   value   = (const float*)d_in[2];
    const float*   abs_q_w = (const float*)d_in[3];
    const float*   abs_k_w = (const float*)d_in[4];
    const float*   rel_w   = (const float*)d_in[5];
    const uint8_t* mask    = (const uint8_t*)d_in[6];

    float* out = (float*)d_out;
    float* p_g = out + OUT_SZ;
    float* p_l = p_g + PG_SZ;

    dim3 block(128);                              // 4 wave32s
    size_t shmem = LDS_FLOATS * sizeof(float);    // ~66.5 KB of the 320 KB WGP LDS

    // 12 global heads: no gate code, no abs weights.
    dim3 gridG(N_CTX / 16, NGLOB, NBATCH);
    attn_mixed_kernel<false><<<gridG, block, shmem, stream>>>(
        query, key, value, abs_q_w, abs_k_w, rel_w, mask, out, p_g);

    // 4 local heads: additive positional weights + sigmoid distance gate.
    dim3 gridL(N_CTX / 16, NLOC, NBATCH);
    attn_mixed_kernel<true><<<gridL, block, shmem, stream>>>(
        query, key, value, abs_q_w, abs_k_w, rel_w, mask, out, p_l);
}
